// down_block_trans_49520972923052
// MI455X (gfx1250) — matrix operand order repaired and checked by reading, not compiled, not run on hardware
//
#include <hip/hip_runtime.h>

typedef __attribute__((ext_vector_type(16))) __bf16 v16bf;
typedef __attribute__((ext_vector_type(8)))  __bf16 v8bf;
typedef __attribute__((ext_vector_type(8)))  float  v8f;
typedef __attribute__((ext_vector_type(4)))  int    v4i;

#define HH   128
#define WW   128
#define HWSZ 16384
#define NBAT 2
#define MTOT (NBAT*HWSZ)   // 32768 pixel rows

// ---------------- CDNA5 async global->LDS staging (ASYNCcnt path) ----------------------
#if __has_builtin(__builtin_amdgcn_global_load_async_to_lds_b128) && __has_builtin(__builtin_amdgcn_s_wait_asynccnt)
#define ASYNC_OK 1
#else
#define ASYNC_OK 0
#endif

static __device__ __forceinline__ void cp16_async(void* lds, const void* glob) {
#if ASYNC_OK
  typedef __attribute__((address_space(1))) v4i gv4i;   // global int4
  typedef __attribute__((address_space(3))) v4i lv4i;   // LDS int4
  __builtin_amdgcn_global_load_async_to_lds_b128((gv4i*)glob, (lv4i*)lds, 0, 0);
#else
  *(v8bf*)lds = *(const v8bf*)glob;
#endif
}
static __device__ __forceinline__ void staging_fence() {
#if ASYNC_OK
  __builtin_amdgcn_s_wait_asynccnt(0);
#endif
  __syncthreads();
}

// ---------------- WMMA fragment loaders (LDS -> VGPR frags, vectorized) ----------------
static __device__ __forceinline__ v16bf frag_cat(v8bf lo, v8bf hi) {
  return __builtin_shufflevector(lo, hi, 0, 1, 2, 3, 4, 5, 6, 7, 8, 9, 10, 11, 12, 13, 14, 15);
}
// A 16x32 bf16: lane[3:0]=M, lane[4]=kh. Per lane the 16 elems are two contiguous runs:
// K = kh*8..kh*8+7 and K = 16+kh*8..+7.  rowbase = &A_lds[row][k-window 0]
static __device__ __forceinline__ v16bf load_a_frag(const __bf16* rowbase, int kh) {
  const __bf16* p = rowbase + kh * 8;
  v8bf lo = *(const v8bf*)p;          // 16B aligned
  v8bf hi = *(const v8bf*)(p + 16);
  return frag_cat(lo, hi);
}
// B 32x16 bf16 from TRANSPOSED LDS tile [N][K]: lane[3:0]=N col, elems K = kh*8..+7 and 16+kh*8..+7
static __device__ __forceinline__ v16bf load_b_fragT(const __bf16* colbase, int kh) {
  const __bf16* p = colbase + kh * 8;
  v8bf lo = *(const v8bf*)p;
  v8bf hi = *(const v8bf*)(p + 16);
  return frag_cat(lo, hi);
}

// ---------------- maxpool 2x2: NCHW f32 (2,64,256,256) -> NHWC f32 (2,16384,64) --------
__global__ void k_maxpool(const float* __restrict__ X, float* __restrict__ P) {
  int idx = blockIdx.x * 256 + threadIdx.x;        // 2*64*128*128 = 2^21
  int x = idx & 127, y = (idx >> 7) & 127, c = (idx >> 14) & 63, b = idx >> 20;
  const float* s = X + (((size_t)(b * 64 + c) * 256 + y * 2) * 256 + x * 2);
  float m = fmaxf(fmaxf(s[0], s[1]), fmaxf(s[256], s[257]));
  P[((size_t)b * HWSZ + y * WW + x) * 64 + c] = m;
}

// ---------------- training-mode BN: per-channel mean / inv-std over N*H*W --------------
__global__ void k_bn_stats(const float* __restrict__ In, int C, float* __restrict__ mean,
                           float* __restrict__ istd) {
  int c = blockIdx.x;
  __shared__ float s1[256], s2[256];
  float a = 0.f, q = 0.f;
  for (int r = threadIdx.x; r < MTOT; r += 256) {
    float v = In[(size_t)r * C + c];
    a += v; q += v * v;
  }
  s1[threadIdx.x] = a; s2[threadIdx.x] = q;
  __syncthreads();
  for (int s = 128; s > 0; s >>= 1) {
    if (threadIdx.x < s) { s1[threadIdx.x] += s1[threadIdx.x + s]; s2[threadIdx.x] += s2[threadIdx.x + s]; }
    __syncthreads();
  }
  if (threadIdx.x == 0) {
    float m = s1[0] * (1.f / MTOT);
    float v = s2[0] * (1.f / MTOT) - m * m;
    mean[c] = m;
    istd[c] = rsqrtf(v + 1e-5f);
  }
}

// ---------------- BN apply (+optional ReLU), writes f32 and/or bf16 NHWC ---------------
__global__ void k_bn_apply(const float* __restrict__ In, const float* __restrict__ mean,
                           const float* __restrict__ istd, const float* __restrict__ g,
                           const float* __restrict__ bb, int C, int relu,
                           float* __restrict__ Of, __bf16* __restrict__ Ob) {
  size_t idx = (size_t)blockIdx.x * 256 + threadIdx.x;   // MTOT*C total, divisible
  int c = (int)(idx & (size_t)(C - 1));                  // C is 64 or 128
  float v = (In[idx] - mean[c]) * istd[c] * g[c] + bb[c];
  if (relu) v = fmaxf(v, 0.f);
  if (Of) Of[idx] = v;
  if (Ob) Ob[idx] = (__bf16)v;
}

// ---------------- weight repacks to bf16 (kept in native [OC][K] = B-transposed) -------
__global__ void k_w1(const float* __restrict__ W, int n, __bf16* __restrict__ BT) {
  int idx = blockIdx.x * 256 + threadIdx.x;     // pure convert: layout already [N][K]
  if (idx >= n) return;
  BT[idx] = (__bf16)W[idx];
}
__global__ void k_w3(const float* __restrict__ W, int OC, int IC, __bf16* __restrict__ BT) {
  int idx = blockIdx.x * 256 + threadIdx.x;     // OC*IC*9 ; W[oc][ic][tap] -> BT[oc][tap*IC+ic]
  if (idx >= OC * IC * 9) return;
  int tap = idx % 9, ic = (idx / 9) % IC, oc = idx / (9 * IC);
  BT[(size_t)oc * 9 * IC + tap * IC + ic] = (__bf16)W[idx];
}

// ---------------- GEMM: C[M,N] = A[M,K]bf16 x BT[N,K]bf16 (+res), f32/bf16 out ---------
// block = 128 thr (4 waves); tile 128(M) x 64(N); per wave 32 rows x 64 cols
__global__ void k_gemm(const __bf16* __restrict__ A, const __bf16* __restrict__ BT,
                       int N, int K, float* __restrict__ Cf, __bf16* __restrict__ Cb,
                       const float* __restrict__ Res) {
  __shared__ alignas(32) __bf16 As[128 * 32];
  __shared__ alignas(32) __bf16 Bs[64 * 32];     // transposed tile [n][k]
  const int tid = threadIdx.x, lane = tid & 31, wave = tid >> 5;
  const int kh = (lane >> 4) & 1;
  const int m0 = blockIdx.x * 128, n0 = blockIdx.y * 64;
  const int cr = tid >> 2, ck = (tid & 3) * 8;   // chunk row / chunk k-offset
  v8f acc[2][4] = {};
  for (int k0 = 0; k0 < K; k0 += 32) {
    {                                            // A tile: 128x32, 4 chunks/thread
      const __bf16* src = &A[(size_t)(m0 + cr) * K + k0 + ck];
      __bf16* dst = &As[cr * 32 + ck];
#pragma unroll
      for (int it = 0; it < 4; ++it) {
        cp16_async(dst, src);
        src += (size_t)32 * K;
        dst += 32 * 32;
      }
    }
    {                                            // B tile: 64x32, 2 chunks/thread
      const __bf16* src = &BT[(size_t)(n0 + cr) * K + k0 + ck];
      __bf16* dst = &Bs[cr * 32 + ck];
#pragma unroll
      for (int it = 0; it < 2; ++it) {
        cp16_async(dst, src);
        src += (size_t)32 * K;
        dst += 32 * 32;
      }
    }
    if (k0 + 32 < K)
      __builtin_prefetch(&A[(size_t)(m0 + tid) * K + k0 + 32], 0, 1);
    staging_fence();
    v16bf a0 = load_a_frag(As + (wave * 32 + (lane & 15)) * 32, kh);
    v16bf a1 = load_a_frag(As + (wave * 32 + 16 + (lane & 15)) * 32, kh);
#pragma unroll
    for (int nt = 0; nt < 4; ++nt) {
      v16bf b = load_b_fragT(Bs + (nt * 16 + (lane & 15)) * 32, kh);
      acc[0][nt] = __builtin_amdgcn_wmma_f32_16x16x32_bf16(false, a0, false, b, (short)0, acc[0][nt], false, false);
      acc[1][nt] = __builtin_amdgcn_wmma_f32_16x16x32_bf16(false, a1, false, b, (short)0, acc[1][nt], false, false);
    }
    __syncthreads();
  }
#pragma unroll
  for (int s = 0; s < 2; ++s)
#pragma unroll
    for (int nt = 0; nt < 4; ++nt) {
      int col = n0 + nt * 16 + (lane & 15);
#pragma unroll
      for (int r = 0; r < 8; ++r) {
        int row = m0 + wave * 32 + s * 16 + ((lane >> 4) << 3) + r;
        size_t idx = (size_t)row * N + col;
        float v = acc[s][nt][r];
        if (Res) v += Res[idx];
        if (Cf) Cf[idx] = v;
        if (Cb) Cb[idx] = (__bf16)v;
      }
    }
}

// ---------------- 3x3-conv implicit GEMM (pad=1), NHWC bf16 in, NHWC f32 out (+res) ----
__global__ void k_conv3(const __bf16* __restrict__ In, const __bf16* __restrict__ BT,
                        int IC, int OC, float* __restrict__ Out, const float* __restrict__ Res) {
  __shared__ alignas(32) __bf16 As[128 * 32];
  __shared__ alignas(32) __bf16 Bs[64 * 32];
  const int tid = threadIdx.x, lane = tid & 31, wave = tid >> 5;
  const int kh = (lane >> 4) & 1;
  const int m0 = blockIdx.x * 128, n0 = blockIdx.y * 64;
  const int Ksz = 9 * IC;
  const int cr = tid >> 2, ck = (tid & 3) * 8;
  v8f acc[2][4] = {};
  for (int k0 = 0; k0 < Ksz; k0 += 32) {
    const int ktap = k0 / IC, ic0 = k0 % IC;     // IC % 32 == 0 -> chunk within one tap
    const int ky = ktap / 3 - 1, kx = ktap % 3 - 1;
    {                                            // A tile with implicit zero padding
      int r = cr;
#pragma unroll
      for (int it = 0; it < 4; ++it, r += 32) {
        int m = m0 + r;
        int b = m >> 14, pix = m & (HWSZ - 1);
        int y = (pix >> 7) + ky, x = (pix & 127) + kx;
        __bf16* dst = &As[r * 32 + ck];
        if ((unsigned)y < (unsigned)HH && (unsigned)x < (unsigned)WW) {
          cp16_async(dst, &In[((size_t)b * HWSZ + y * WW + x) * IC + ic0 + ck]);
        } else {
          v8bf z = {};
          *(v8bf*)dst = z;
        }
      }
    }
    {
      const __bf16* src = &BT[(size_t)(n0 + cr) * Ksz + k0 + ck];
      __bf16* dst = &Bs[cr * 32 + ck];
#pragma unroll
      for (int it = 0; it < 2; ++it) {
        cp16_async(dst, src);
        src += (size_t)32 * Ksz;
        dst += 32 * 32;
      }
    }
    staging_fence();
    v16bf a0 = load_a_frag(As + (wave * 32 + (lane & 15)) * 32, kh);
    v16bf a1 = load_a_frag(As + (wave * 32 + 16 + (lane & 15)) * 32, kh);
#pragma unroll
    for (int nt = 0; nt < 4; ++nt) {
      v16bf b = load_b_fragT(Bs + (nt * 16 + (lane & 15)) * 32, kh);
      acc[0][nt] = __builtin_amdgcn_wmma_f32_16x16x32_bf16(false, a0, false, b, (short)0, acc[0][nt], false, false);
      acc[1][nt] = __builtin_amdgcn_wmma_f32_16x16x32_bf16(false, a1, false, b, (short)0, acc[1][nt], false, false);
    }
    __syncthreads();
  }
#pragma unroll
  for (int s = 0; s < 2; ++s)
#pragma unroll
    for (int nt = 0; nt < 4; ++nt) {
      int col = n0 + nt * 16 + (lane & 15);
#pragma unroll
      for (int r = 0; r < 8; ++r) {
        int row = m0 + wave * 32 + s * 16 + ((lane >> 4) << 3) + r;
        size_t idx = (size_t)row * OC + col;
        float v = acc[s][nt][r];
        if (Res) v += Res[idx];
        Out[idx] = v;
      }
    }
}

// ---------------- depthwise 3x3 (pad=1), NHWC f32 in -> NHWC bf16 out -------------------
__global__ void k_dw3(const float* __restrict__ In, const float* __restrict__ Wd,
                      __bf16* __restrict__ Ob) {
  int idx = blockIdx.x * 256 + threadIdx.x;       // 2*16384*128 = 2^22
  int c = idx & 127, pix = (idx >> 7) & (HWSZ - 1), b = idx >> 21;
  int y = pix >> 7, x = pix & 127;
  float s = 0.f;
#pragma unroll
  for (int ky = 0; ky < 3; ++ky)
#pragma unroll
    for (int kx = 0; kx < 3; ++kx) {
      int yy = y + ky - 1, xx = x + kx - 1;
      if ((unsigned)yy < (unsigned)HH && (unsigned)xx < (unsigned)WW)
        s += Wd[c * 9 + ky * 3 + kx] * In[(((size_t)b * HWSZ + yy * WW + xx) << 7) + c];
    }
  Ob[idx] = (__bf16)s;
}

// ---------------- bilinear (align_corners) downsample of k,v to 16x16, head-split ------
// qkv NHWC bf16 [B][HW][384]; KdsT[bh][j(256)][d(32)]  (B-transposed for score GEMM)
//                            VdsT[bh][d(32)][j(256)]   (B-transposed for PV GEMM)
__global__ void k_downsample(const __bf16* __restrict__ QKV, __bf16* __restrict__ KdsT,
                             __bf16* __restrict__ VdsT) {
  int pos = blockIdx.x;                  // 2*16*16 = 512
  int c = threadIdx.x;                   // 128 channels
  int b = pos >> 8, r = (pos >> 4) & 15, s = pos & 15;
  float ph = r * (127.f / 15.f); int lh = min((int)floorf(ph), 126); float fh = ph - lh;
  float pw = s * (127.f / 15.f); int lw = min((int)floorf(pw), 126); float fw = pw - lw;
  float w00 = (1.f - fh) * (1.f - fw), w01 = (1.f - fh) * fw, w10 = fh * (1.f - fw), w11 = fh * fw;
  auto samp = [&](int ch, int y, int x) -> float {
    return (float)QKV[((size_t)b * HWSZ + y * WW + x) * 384 + ch];
  };
  float kv = w00 * samp(128 + c, lh, lw) + w01 * samp(128 + c, lh, lw + 1)
           + w10 * samp(128 + c, lh + 1, lw) + w11 * samp(128 + c, lh + 1, lw + 1);
  float vv = w00 * samp(256 + c, lh, lw) + w01 * samp(256 + c, lh, lw + 1)
           + w10 * samp(256 + c, lh + 1, lw) + w11 * samp(256 + c, lh + 1, lw + 1);
  int d = c >> 2, h = c & 3;             // channel = d*HEADS + h
  int j = r * 16 + s;
  KdsT[((size_t)(b * 4 + h) * 256 + j) * 32 + d] = (__bf16)kv;
  VdsT[((size_t)(b * 4 + h) * 32 + d) * 256 + j] = (__bf16)vv;
}

// ---------------- fused linear attention: scores + rel-bias + softmax + P*V ------------
// grid.x = 256 (64-row i tiles), grid.y = 8 (b*4+h); block = 128 threads (4 waves)
__global__ void k_attention(const __bf16* __restrict__ QKV, const __bf16* __restrict__ KdsT,
                            const __bf16* __restrict__ VdsT, const float* __restrict__ table,
                            float* __restrict__ AttnOut) {
  __shared__ alignas(32) __bf16 Qs[64 * 32];     //  4 KB   [i][d]
  __shared__ alignas(32) __bf16 Ks[256 * 32];    // 16 KB   [j][d]  (B-transposed)
  __shared__ alignas(32) __bf16 Vs[32 * 256];    // 16 KB   [d][j]  (B-transposed)
  __shared__ alignas(32) float  Ss[64 * 256];    // 64 KB  (P bf16 overlaid in place)
  const int bh = blockIdx.y, b = bh >> 2, h = bh & 3;
  const int i0 = blockIdx.x * 64;
  const int tid = threadIdx.x, lane = tid & 31, wave = tid >> 5;
  const int kh = (lane >> 4) & 1;

  {                                              // K/V: contiguous async copies
    const __bf16* ksrc = KdsT + (size_t)bh * 8192 + tid * 8;
    const __bf16* vsrc = VdsT + (size_t)bh * 8192 + tid * 8;
#pragma unroll
    for (int it = 0; it < 8; ++it) {
      cp16_async(&Ks[tid * 8 + it * 1024], ksrc + it * 1024);
      cp16_async(&Vs[tid * 8 + it * 1024], vsrc + it * 1024);
    }
  }
  for (int t = tid; t < 64 * 32; t += 128) {     // strided head gather (channel = d*4+h)
    int r = t >> 5, d = t & 31;
    Qs[t] = QKV[((size_t)b * HWSZ + (i0 + r)) * 384 + (d * 4 + h)];
  }
  staging_fence();

  { // scores: one WMMA per 16x16 j-tile (K = dim_head = 32 exactly)
    v16bf a = load_a_frag(Qs + ((wave << 4) + (lane & 15)) * 32, kh);
    for (int jt = 0; jt < 16; ++jt) {
      v16bf bfr = load_b_fragT(Ks + (jt * 16 + (lane & 15)) * 32, kh);
      v8f z = {};
      v8f s = __builtin_amdgcn_wmma_f32_16x16x32_bf16(false, a, false, bfr, (short)0, z, false, false);
#pragma unroll
      for (int r = 0; r < 8; ++r)
        Ss[((wave << 4) + (kh << 3) + r) * 256 + jt * 16 + (lane & 15)] = s[r];
    }
  }
  __syncthreads();

  const float scale = 0.17677669529663687f;      // 32^-0.5
  for (int t = tid; t < 64 * 256; t += 128) {
    int r = t >> 8, j = t & 255;
    int pix = i0 + r;
    int ih = (pix >> 7) >> 3, iw = (pix & 127) >> 3;
    int jh = j >> 4, jw = j & 15;
    int idx = (ih - jh + 15) * 31 + (iw - jw + 15);
    Ss[t] = (Ss[t] + table[idx * 4 + h]) * scale;
  }
  __syncthreads();

  __bf16* Psb = (__bf16*)Ss;                     // bf16 P row r in first half of f32 row r
  if (tid < 64) {
    float mx = -1e30f;
    for (int j = 0; j < 256; ++j) mx = fmaxf(mx, Ss[tid * 256 + j]);
    float sum = 0.f;
    for (int j = 0; j < 256; ++j) { float e = __expf(Ss[tid * 256 + j] - mx); Ss[tid * 256 + j] = e; sum += e; }
    float inv = 1.f / sum;
    for (int j = 0; j < 256; ++j)                // read f32 @4j then write bf16 @2j: safe ascending
      Psb[tid * 512 + j] = (__bf16)(Ss[tid * 256 + j] * inv);
  }
  __syncthreads();

  v8f acc[2] = {};
  for (int ks = 0; ks < 8; ++ks) {               // O = P(64x256) x V(256x32)
    v16bf a = load_a_frag(Psb + ((wave << 4) + (lane & 15)) * 512 + ks * 32, kh);
#pragma unroll
    for (int nt = 0; nt < 2; ++nt) {
      v16bf bfr = load_b_fragT(Vs + (nt * 16 + (lane & 15)) * 256 + ks * 32, kh);
      acc[nt] = __builtin_amdgcn_wmma_f32_16x16x32_bf16(false, a, false, bfr, (short)0, acc[nt], false, false);
    }
  }
#pragma unroll
  for (int nt = 0; nt < 2; ++nt) {
    int d = nt * 16 + (lane & 15);
#pragma unroll
    for (int r = 0; r < 8; ++r) {
      int row = i0 + (wave << 4) + ((lane >> 4) << 3) + r;
      AttnOut[((size_t)b * HWSZ + row) * 128 + d * 4 + h] = acc[nt][r];
    }
  }
}

// ---------------- NHWC f32 -> NCHW f32 final output ------------------------------------
__global__ void k_to_nchw(const float* __restrict__ In, float* __restrict__ Out) {
  int idx = blockIdx.x * 256 + threadIdx.x;      // 2*128*16384 = 2^22
  int pix = idx & (HWSZ - 1), c = (idx >> 14) & 127, b = idx >> 21;
  Out[idx] = In[((size_t)b * HWSZ + pix) * 128 + c];
}

// =======================================================================================
extern "C" void kernel_launch(void* const* d_in, const int* in_sizes, int n_in,
                              void* d_out, int out_size, void* d_ws, size_t ws_size,
                              hipStream_t stream) {
  const float* x        = (const float*)d_in[0];
  const float* bb_bn1_g = (const float*)d_in[1];
  const float* bb_bn1_b = (const float*)d_in[2];
  const float* bb_conv1 = (const float*)d_in[3];
  const float* bb_bn2_g = (const float*)d_in[4];
  const float* bb_bn2_b = (const float*)d_in[5];
  const float* bb_conv2 = (const float*)d_in[6];
  const float* bb_sbn_g = (const float*)d_in[7];
  const float* bb_sbn_b = (const float*)d_in[8];
  const float* bb_sconv = (const float*)d_in[9];
  const float* tb_bn1_g = (const float*)d_in[10];
  const float* tb_bn1_b = (const float*)d_in[11];
  const float* tb_dwqkv = (const float*)d_in[12];
  const float* tb_pwqkv = (const float*)d_in[13];
  const float* tb_dwout = (const float*)d_in[14];
  const float* tb_pwout = (const float*)d_in[15];
  const float* tb_rel   = (const float*)d_in[16];
  const float* tb_bn2_g = (const float*)d_in[17];
  const float* tb_bn2_b = (const float*)d_in[18];
  const float* tb_mlp   = (const float*)d_in[19];

  size_t off = 0;
  auto alloc = [&](size_t bytes) {
    void* p = (char*)d_ws + off;
    off += (bytes + 255) & ~(size_t)255;
    return p;
  };
  float*  p_buf   = (float*)alloc((size_t)MTOT * 64 * 4);     // pooled, NHWC
  float*  h1      = (float*)alloc((size_t)MTOT * 128 * 4);
  float*  sc      = (float*)alloc((size_t)MTOT * 128 * 4);
  float*  outb    = (float*)alloc((size_t)MTOT * 128 * 4);
  float*  tmp     = (float*)alloc((size_t)MTOT * 128 * 4);
  float*  attnout = (float*)alloc((size_t)MTOT * 128 * 4);
  __bf16* act     = (__bf16*)alloc((size_t)MTOT * 128 * 2);
  __bf16* act2    = (__bf16*)alloc((size_t)MTOT * 128 * 2);
  __bf16* qkvb    = (__bf16*)alloc((size_t)MTOT * 384 * 2);
  __bf16* kdsT    = (__bf16*)alloc((size_t)8 * 256 * 32 * 2);
  __bf16* vdsT    = (__bf16*)alloc((size_t)8 * 32 * 256 * 2);
  __bf16* wc1     = (__bf16*)alloc((size_t)128 * 576 * 2);    // [OC][9*IC]
  __bf16* wc2     = (__bf16*)alloc((size_t)128 * 1152 * 2);
  __bf16* wsc     = (__bf16*)alloc((size_t)128 * 64 * 2);     // [OC][IC] (native)
  __bf16* wpwqkv  = (__bf16*)alloc((size_t)384 * 128 * 2);
  __bf16* wpwout  = (__bf16*)alloc((size_t)128 * 128 * 2);
  __bf16* wmlp    = (__bf16*)alloc((size_t)128 * 128 * 2);
  float*  meanb   = (float*)alloc(128 * 4);
  float*  istdb   = (float*)alloc(128 * 4);
  (void)ws_size; (void)n_in; (void)in_sizes; (void)out_size;

  // weight repacks (B kept transposed [N][K] -> contiguous frag loads)
  k_w3<<<(128 * 64 * 9 + 255) / 256, 256, 0, stream>>>(bb_conv1, 128, 64, wc1);
  k_w3<<<(128 * 128 * 9 + 255) / 256, 256, 0, stream>>>(bb_conv2, 128, 128, wc2);
  k_w1<<<(128 * 64 + 255) / 256, 256, 0, stream>>>(bb_sconv, 128 * 64, wsc);

  // maxpool
  k_maxpool<<<(MTOT * 64) / 256, 256, 0, stream>>>(x, p_buf);

  // BasicBlock
  k_bn_stats<<<64, 256, 0, stream>>>(p_buf, 64, meanb, istdb);
  k_bn_apply<<<(MTOT * 64) / 256, 256, 0, stream>>>(p_buf, meanb, istdb, bb_bn1_g, bb_bn1_b, 64, 1, nullptr, act);
  k_conv3<<<dim3(MTOT / 128, 2), 128, 0, stream>>>(act, wc1, 64, 128, h1, nullptr);
  k_bn_stats<<<128, 256, 0, stream>>>(h1, 128, meanb, istdb);
  k_bn_apply<<<(MTOT * 128) / 256, 256, 0, stream>>>(h1, meanb, istdb, bb_bn2_g, bb_bn2_b, 128, 1, nullptr, act2);
  k_bn_stats<<<64, 256, 0, stream>>>(p_buf, 64, meanb, istdb);
  k_bn_apply<<<(MTOT * 64) / 256, 256, 0, stream>>>(p_buf, meanb, istdb, bb_sbn_g, bb_sbn_b, 64, 1, nullptr, act);
  k_gemm<<<dim3(MTOT / 128, 2), 128, 0, stream>>>(act, wsc, 128, 64, sc, nullptr, nullptr);
  k_conv3<<<dim3(MTOT / 128, 2), 128, 0, stream>>>(act2, wc2, 128, 128, outb, sc);

  // BasicTransBlocks
  for (int i = 0; i < 2; ++i) {
    k_w1<<<(384 * 128 + 255) / 256, 256, 0, stream>>>(tb_pwqkv + (size_t)i * 384 * 128, 384 * 128, wpwqkv);
    k_w1<<<(128 * 128 + 255) / 256, 256, 0, stream>>>(tb_pwout + (size_t)i * 128 * 128, 128 * 128, wpwout);
    k_w1<<<(128 * 128 + 255) / 256, 256, 0, stream>>>(tb_mlp + (size_t)i * 128 * 128, 128 * 128, wmlp);

    // attention branch
    k_bn_stats<<<128, 256, 0, stream>>>(outb, 128, meanb, istdb);
    k_bn_apply<<<(MTOT * 128) / 256, 256, 0, stream>>>(outb, meanb, istdb,
        tb_bn1_g + i * 128, tb_bn1_b + i * 128, 128, 0, tmp, nullptr);
    k_dw3<<<(MTOT * 128) / 256, 256, 0, stream>>>(tmp, tb_dwqkv + (size_t)i * 128 * 9, act);
    k_gemm<<<dim3(MTOT / 128, 6), 128, 0, stream>>>(act, wpwqkv, 384, 128, nullptr, qkvb, nullptr);
    k_downsample<<<512, 128, 0, stream>>>(qkvb, kdsT, vdsT);
    k_attention<<<dim3(HWSZ / 64, 8), 128, 0, stream>>>(qkvb, kdsT, vdsT, tb_rel + (size_t)i * 961 * 4, attnout);
    k_dw3<<<(MTOT * 128) / 256, 256, 0, stream>>>(attnout, tb_dwout + (size_t)i * 128 * 9, act);
    k_gemm<<<dim3(MTOT / 128, 2), 128, 0, stream>>>(act, wpwout, 128, 128, outb, nullptr, outb); // out = a + out

    // MLP branch
    k_bn_stats<<<128, 256, 0, stream>>>(outb, 128, meanb, istdb);
    k_bn_apply<<<(MTOT * 128) / 256, 256, 0, stream>>>(outb, meanb, istdb,
        tb_bn2_g + i * 128, tb_bn2_b + i * 128, 128, 1, nullptr, act);
    k_gemm<<<dim3(MTOT / 128, 2), 128, 0, stream>>>(act, wmlp, 128, 128, outb, nullptr, outb);   // out = h + out
  }

  k_to_nchw<<<(MTOT * 128) / 256, 256, 0, stream>>>(outb, (float*)d_out);
}
